// PCRNNModel_78048145703039
// MI455X (gfx1250) — matrix operand order repaired and checked by reading, not compiled, not run on hardware
//
#include <hip/hip_runtime.h>
#include <hip/hip_bf16.h>

// ---------------------------------------------------------------------------
// PC-RNN scan for MI455X (gfx1250, wave32, WMMA).
//   B=32,T=512,F=128,U=256,L=2.  grid=2 blocks (16 batch rows each),
//   block=256 threads (8 wave32s), persistent over the whole T loop.
//   Latency-optimized: all state-dependent GEMM halves issue at step top
//   (7 independent 8-deep WMMA chains), only resid halves sit on the true
//   dependency path; gates replicated per-wave (register moving-stats +
//   __shfl broadcast); loss buffers double-buffered. 3 barriers/step.
// ---------------------------------------------------------------------------

typedef __attribute__((ext_vector_type(16))) _Float16 v16h;
typedef __attribute__((ext_vector_type(8)))  float    v8f;

#define B_  32
#define T_  512
#define F_  128
#define U_  256
#define D_  1668          // (U+3F+2) + (4U+2)
#define L1OFF 642         // start of layer-1 chunk in D

// f16 B-fragment workspace layout (element offsets)
#define WP0_OFF 0         // 8 ntiles * 8  ktiles * 512
#define WT0_OFF 32768     // 16 ntiles * 12 ktiles * 512
#define WP1_OFF 131072    // 16 ntiles * 8  ktiles * 512
#define WT1_OFF 196608    // 16 ntiles * 16 ktiles * 512
#define FRAG_TOT 327680

// A-fragment address for activation value at (row m, k-position u).
// 16-bit A 16x32 layout (ISA 7.12.2): lane = m + 16*half, half = (k%32)>>3 & 1,
// j = (k&7) + 8*(k%32 >= 16).
__device__ __forceinline__ int afrag_idx(int m, int u) {
    int kt    = u >> 5;
    int kl    = u & 31;
    int halfk = (kl >> 3) & 1;
    int j     = (kl & 7) | (((kl >> 4) & 1) << 3);
    return (kt * 32 + (m + (halfk << 4))) * 16 + j;
}

// error_gate_movingMV (online mean/var -> clipped tanh z-score gate)
__device__ __forceinline__ void moving_gate(float& stp, float& m1, float& m2,
                                            float val, float gb, float& gate_out) {
    float stp_prev = stp, m1_prev = m1, m2_prev = m2;
    float s1  = stp + 1.0f;
    float m1o = m1;
    float m1n = m1o + (val - m1o) / s1;
    float m2n = m2 + (val - m1n) * (val - m1o);
    float sd  = fmaxf(sqrtf(m2n / s1 + 1e-8f), 1e-8f);
    float g   = tanhf(fmaxf((val - m1n) / sd, 0.0f));
    float filt = 1.0f - g;
    float s2   = s1 * filt;
    float m2f  = m2n * filt;
    float gbc  = 1.0f - gb;
    stp = s2  * gb + stp_prev * gbc;
    m1  = m1n * gb + m1_prev  * gbc;
    m2  = m2f * gb + m2_prev  * gbc;
    gate_out = g * gb;
}

// ---------------------------------------------------------------------------
// Prep: swizzle f32 weights [K,N] (row-major) into f16 B-fragments.
// B 32x16 16-bit layout: lane L holds col n = nt*16 + (L&15); element j of the
// per-lane v16h is k = kt*32 + 8*(L>>4) + (j&7) + 16*(j>>3).
// ---------------------------------------------------------------------------
__global__ void prep_frags(const float* __restrict__ Wp0, const float* __restrict__ Wt0,
                           const float* __restrict__ Wp1, const float* __restrict__ Wt1,
                           _Float16* __restrict__ frag) {
    int e = blockIdx.x * blockDim.x + threadIdx.x;
    if (e >= FRAG_TOT) return;
    const float* W; int KT, N, le;
    if (e < WT0_OFF)       { W = Wp0; KT = 8;  N = F_; le = e; }
    else if (e < WP1_OFF)  { W = Wt0; KT = 12; N = U_; le = e - WT0_OFF; }
    else if (e < WT1_OFF)  { W = Wp1; KT = 8;  N = U_; le = e - WP1_OFF; }
    else                   { W = Wt1; KT = 16; N = U_; le = e - WT1_OFF; }
    int j    = le & 15;
    int lane = (le >> 4) & 31;
    int tile = le >> 9;
    int kt   = tile % KT;
    int nt   = tile / KT;
    int k = kt * 32 + 8 * (lane >> 4) + (j & 7) + 16 * (j >> 3);
    int n = nt * 16 + (lane & 15);
    frag[e] = (_Float16)W[(size_t)k * N + n];
}

// ---------------------------------------------------------------------------
// Persistent scan kernel.
// ---------------------------------------------------------------------------
__global__ __launch_bounds__(256, 1)
void pcrnn_scan(const float* __restrict__ x,
                const float* __restrict__ bp0, const float* __restrict__ bt0,
                const float* __restrict__ bp1, const float* __restrict__ bt1,
                const _Float16* __restrict__ frags,
                float* __restrict__ out) {
    const _Float16* Wp0f = frags + WP0_OFF;
    const _Float16* Wt0f = frags + WT0_OFF;
    const _Float16* Wp1f = frags + WP1_OFF;
    const _Float16* Wt1f = frags + WT1_OFF;

    // A-fragment activation buffers (f16): A0 = [st0 (kt0..7) | resid0 (kt8..11)]
    //                                      A1 = [st1 (kt0..7) | resid1 (kt8..15)]
    __shared__ __align__(32) _Float16 A0[12 * 512];
    __shared__ __align__(32) _Float16 A1[16 * 512];
    __shared__ float loss0[2][16], loss1[2][16];   // double-buffered by t parity

    const int tid  = threadIdx.x;
    const int w    = tid >> 5;        // wave id 0..7
    const int lane = tid & 31;
    const int hf   = lane >> 4;       // lane half
    const int ln   = lane & 15;       // col within tile / replicated gate row
    const int bbase = blockIdx.x * 16;

    for (int i = tid; i < 12 * 512; i += 256) A0[i] = (_Float16)0.0f;
    for (int i = tid; i < 16 * 512; i += 256) A1[i] = (_Float16)0.0f;
    if (tid < 16) {
        loss0[0][tid] = loss0[1][tid] = 0.0f;
        loss1[0][tid] = loss1[1][tid] = 0.0f;
    }
    // Per-wave replicated moving-stat state: lane ln owns row ln (halves dup).
    float mg0s = 0.0f, mg0m1 = 0.0f, mg0m2 = 0.0f;
    float mg1s = 0.0f, mg1m1 = 0.0f, mg1m2 = 0.0f;
    v8f p1reg[2]; p1reg[0] = (v8f){}; p1reg[1] = (v8f){};   // prev-step preds1 (topdown)
    float s0reg[2][8];                                      // this-step state0
    __syncthreads();

    for (int t = 0; t < T_; ++t) {
        const int p = t & 1;

        // ============ Top-of-step: every GEMM half that only needs the
        // previous-step state issues now -> 7 independent 8-deep WMMA chains.
        v8f accP0 = {};
        v8f accP1[2]; accP1[0] = (v8f){}; accP1[1] = (v8f){};
        v8f accT0[2]; accT0[0] = (v8f){}; accT0[1] = (v8f){};
        v8f accT1[2]; accT1[0] = (v8f){}; accT1[1] = (v8f){};
#pragma unroll 4
        for (int kt = 0; kt < 8; ++kt) {
            v16h a0 = *(const v16h*)&A0[(kt * 32 + lane) * 16];
            v16h a1 = *(const v16h*)&A1[(kt * 32 + lane) * 16];
            v16h b = *(const v16h*)&Wp0f[((w * 8 + kt) * 32 + lane) * 16];
            accP0 = __builtin_amdgcn_wmma_f32_16x16x32_f16(false, a0, false, b,
                                                           (short)0, accP0, false, false);
#pragma unroll
            for (int tt = 0; tt < 2; ++tt) {
                int nt = 2 * w + tt;
                v16h bt0f = *(const v16h*)&Wt0f[(((size_t)nt * 12 + kt) * 32 + lane) * 16];
                accT0[tt] = __builtin_amdgcn_wmma_f32_16x16x32_f16(false, a0, false, bt0f,
                                                                   (short)0, accT0[tt], false, false);
                v16h bp1f = *(const v16h*)&Wp1f[(((size_t)nt * 8 + kt) * 32 + lane) * 16];
                accP1[tt] = __builtin_amdgcn_wmma_f32_16x16x32_f16(false, a1, false, bp1f,
                                                                   (short)0, accP1[tt], false, false);
                v16h bt1f = *(const v16h*)&Wt1f[(((size_t)nt * 16 + kt) * 32 + lane) * 16];
                accT1[tt] = __builtin_amdgcn_wmma_f32_16x16x32_f16(false, a1, false, bt1f,
                                                                   (short)0, accT1[tt], false, false);
            }
        }
        // Prefetch the resid-half weight fragments (consumed after barriers).
#pragma unroll
        for (int tt = 0; tt < 2; ++tt) {
            int nt = 2 * w + tt;
#pragma unroll
            for (int kt = 8; kt < 12; ++kt)
                __builtin_prefetch(&Wt0f[(((size_t)nt * 12 + kt) * 32 + lane) * 16], 0, 3);
#pragma unroll
            for (int kt = 8; kt < 16; ++kt)
                __builtin_prefetch(&Wt1f[(((size_t)nt * 16 + kt) * 32 + lane) * 16], 0, 3);
        }

        // ============ Phase 1 elementwise: resid0 / err0 / loss0 ============
        {
            int f = w * 16 + ln;
            float bpv = bp0[f];
#pragma unroll
            for (int r = 0; r < 8; ++r) {
                int m = r + hf * 8;
                size_t b_ = (size_t)(bbase + m);
                size_t xoff = (b_ * T_ + t) * F_ + f;
                float xv = x[xoff];
                if (t + 1 < T_) __builtin_prefetch(&x[xoff + F_], 0, 3);
                float pred  = accP0[r] + bpv;
                float resid = xv - pred;
                float err   = resid * resid;
                size_t ob = (b_ * T_ + (size_t)t) * D_;
                out[ob + 256 + f] = err;
                out[ob + 384 + f] = pred;
                out[ob + 512 + f] = resid;
                A0[afrag_idx(m, 256 + f)] = (_Float16)resid;
                float ls = err;
#pragma unroll
                for (int mk = 1; mk < 16; mk <<= 1) ls += __shfl_xor(ls, mk, 32);
                if (ln == 0) atomicAdd(&loss0[p][m], ls);
            }
        }
        __syncthreads();   // [#A] loss0 complete; A0 resid visible; all A0/A1 reads done

        // ============ Gate 0 (replicated on every wave; lane -> row ln) =====
        float lv0 = loss0[p][ln] * (1.0f / F_);
        float g0;
        moving_gate(mg0s, mg0m1, mg0m2, lv0, 1.0f, g0);
        if (tid < 16) {
            size_t ob = ((size_t)(bbase + tid) * T_ + t) * D_;
            out[ob + 640] = lv0;
            out[ob + 641] = g0;
            loss0[1 - p][tid] = 0.0f;      // pre-zero next step's buffers
            loss1[1 - p][tid] = 0.0f;
        }

        // ============ Phase 2: finish trans0, state0, store new st0 =========
#pragma unroll
        for (int tt = 0; tt < 2; ++tt) {
            int nt = 2 * w + tt;
            v8f acc = accT0[tt];
#pragma unroll
            for (int kt = 8; kt < 12; ++kt) {
                v16h a = *(const v16h*)&A0[(kt * 32 + lane) * 16];
                v16h b = *(const v16h*)&Wt0f[(((size_t)nt * 12 + kt) * 32 + lane) * 16];
                acc = __builtin_amdgcn_wmma_f32_16x16x32_f16(false, a, false, b,
                                                             (short)0, acc, false, false);
            }
            int u = nt * 16 + ln;
            float btv = bt0[u];
#pragma unroll
            for (int r = 0; r < 8; ++r) {
                int m = r + hf * 8;
                float gm = __shfl(g0, m, 32);
                float sp = acc[r] + btv;
                float s  = tanhf(gm * p1reg[tt][r] + (1.0f - gm) * sp);
                s0reg[tt][r] = s;
                out[((size_t)(bbase + m) * T_ + t) * D_ + u] = s;
                A0[afrag_idx(m, u)] = (_Float16)s;          // new st0 (all readers past #A)
            }
        }

        // ============ Phase 3 elementwise: resid1 / err1 / loss1 ============
#pragma unroll
        for (int tt = 0; tt < 2; ++tt) {
            int u = (2 * w + tt) * 16 + ln;
            float bpv = bp1[u];
#pragma unroll
            for (int r = 0; r < 8; ++r) {
                int m = r + hf * 8;
                float pv    = accP1[tt][r] + bpv;
                float resid = s0reg[tt][r] - pv;
                float err   = resid * resid;
                size_t ob = ((size_t)(bbase + m) * T_ + t) * D_ + L1OFF;
                out[ob + 256 + u] = err;
                out[ob + 512 + u] = pv;
                out[ob + 768 + u] = resid;
                A1[afrag_idx(m, 256 + u)] = (_Float16)resid;
                p1reg[tt][r] = pv;                 // topdown for next step
                float ls = err;
#pragma unroll
                for (int mk = 1; mk < 16; mk <<= 1) ls += __shfl_xor(ls, mk, 32);
                if (ln == 0) atomicAdd(&loss1[p][m], ls);
            }
        }
        __syncthreads();   // [#B] loss1 complete; A1 resid visible

        // ============ Gate 1 (replicated; gate_below = g0) ==================
        float lv1 = loss1[p][ln] * (1.0f / U_);
        float g1;
        moving_gate(mg1s, mg1m1, mg1m2, lv1, g0, g1);
        if (tid < 16) {
            size_t ob = ((size_t)(bbase + tid) * T_ + t) * D_ + L1OFF;
            out[ob + 1024] = lv1;
            out[ob + 1025] = g1;
        }

        // ============ Phase 4: finish trans1 (2 split chains), new st1 ======
#pragma unroll
        for (int tt = 0; tt < 2; ++tt) {
            int nt = 2 * w + tt;
            v8f accA = accT1[tt];
            v8f accB = {};
#pragma unroll
            for (int kt = 8; kt < 12; ++kt) {
                v16h a = *(const v16h*)&A1[(kt * 32 + lane) * 16];
                v16h b = *(const v16h*)&Wt1f[(((size_t)nt * 16 + kt) * 32 + lane) * 16];
                accA = __builtin_amdgcn_wmma_f32_16x16x32_f16(false, a, false, b,
                                                              (short)0, accA, false, false);
            }
#pragma unroll
            for (int kt = 12; kt < 16; ++kt) {
                v16h a = *(const v16h*)&A1[(kt * 32 + lane) * 16];
                v16h b = *(const v16h*)&Wt1f[(((size_t)nt * 16 + kt) * 32 + lane) * 16];
                accB = __builtin_amdgcn_wmma_f32_16x16x32_f16(false, a, false, b,
                                                              (short)0, accB, false, false);
            }
            v8f acc = accA + accB;
            int u = nt * 16 + ln;
            float btv = bt1[u];
#pragma unroll
            for (int r = 0; r < 8; ++r) {
                int m  = r + hf * 8;
                float gbm = __shfl(g0, m, 32);
                float s   = tanhf(acc[r] + btv);
                int ai = afrag_idx(m, u);          // owner lane: safe read+write
                float old = (float)A1[ai];
                s = gbm * s + (1.0f - gbm) * old;
                A1[ai] = (_Float16)s;
                out[((size_t)(bbase + m) * T_ + t) * D_ + L1OFF + u] = s;
            }
        }
        __syncthreads();   // [#C] new st0/st1 visible for next step's top GEMMs
    }
}

extern "C" void kernel_launch(void* const* d_in, const int* in_sizes, int n_in,
                              void* d_out, int out_size, void* d_ws, size_t ws_size,
                              hipStream_t stream) {
    const float* x   = (const float*)d_in[0];
    const float* Wp0 = (const float*)d_in[1];
    const float* bp0 = (const float*)d_in[2];
    const float* Wt0 = (const float*)d_in[3];
    const float* bt0 = (const float*)d_in[4];
    const float* Wp1 = (const float*)d_in[5];
    const float* bp1 = (const float*)d_in[6];
    const float* Wt1 = (const float*)d_in[7];
    const float* bt1 = (const float*)d_in[8];
    _Float16* frags = (_Float16*)d_ws;

    prep_frags<<<(FRAG_TOT + 255) / 256, 256, 0, stream>>>(Wp0, Wt0, Wp1, Wt1, frags);
    pcrnn_scan<<<2, 256, 0, stream>>>(x, bp0, bt0, bp1, bt1, frags, (float*)d_out);
}
